// VQVAE_57337813401661
// MI455X (gfx1250) — matrix operand order repaired and checked by reading, not compiled, not run on hardware
//
#include <hip/hip_runtime.h>
#include <stdint.h>

// ---------------------------------------------------------------------------
// VQ-VAE forward for MI455X (gfx1250).
//  * f16 activations/weights in global (halves HBM traffic; L2-resident),
//    f32 WMMA accumulation: v_wmma_f32_16x16x32_f16.
//  * Weight tiles staged LDS via Tensor Data Mover (tensor_load_to_lds with
//    hardware row padding -> bank-conflict-free fragment reads), overlapped
//    with the im2col gather; s_wait_tensorcnt + block barrier to consume.
//  * All integer div/mod hoisted out of the k-loop into per-block /
//    per-k-step LUTs.
// ---------------------------------------------------------------------------

typedef _Float16 v8h  __attribute__((ext_vector_type(8)));
typedef _Float16 v16h __attribute__((ext_vector_type(16)));
typedef float    v8f  __attribute__((ext_vector_type(8)));
typedef unsigned int tdm4u __attribute__((ext_vector_type(4)));
typedef int          tdm8i __attribute__((ext_vector_type(8)));
typedef int          tdm4i __attribute__((ext_vector_type(4)));

#define FLAG_RELU_IN   1
#define FLAG_RELU_OUT  2
#define FLAG_TANH_OUT  4
#define FLAG_DECONV    8

#define LPAD 40   // conv LDS row stride (halves): 32 data + 8 pad (80B rows)
#define CPAD 72   // vq  LDS row stride (halves): 64 data + 8 pad (144B rows)

#if __has_builtin(__builtin_amdgcn_tensor_load_to_lds) && __has_builtin(__builtin_amdgcn_s_wait_tensorcnt)
#define USE_TDM 1
#else
#define USE_TDM 0
#endif

static __device__ __forceinline__ v16h cat8(v8h lo, v8h hi) {
  return __builtin_shufflevector(lo, hi, 0,1,2,3,4,5,6,7,8,9,10,11,12,13,14,15);
}

#if USE_TDM
// 2D f16 tile -> LDS via TDM. dim0/rows are the *remaining* tensor extent from
// the tile origin (OOB reads zero-fill the tile edge). LDS row padding:
// pad_amount(+1) DWORDs inserted after every 2^(pad_interval+1) DWORDs.
static __device__ __forceinline__ void tdm_load_2d_f16(
    const _Float16* g, unsigned lds_off, int dim0, int rows,
    int tile_d0, int tile_rows, int stride0, int pad_iv, int pad_amt)
{
  unsigned long long ga = (unsigned long long)(uintptr_t)g;
  tdm4u g0;
  g0.x = 1u;                                   // count=1, no gather
  g0.y = lds_off;                              // LDS byte address
  g0.z = (unsigned)ga;                         // global_addr[31:0]
  g0.w = (unsigned)((ga >> 32) & 0x01FFFFFFull) | 0x80000000u; // addr[56:32] | type=2
  tdm8i g1;
  g1[0] = (1 << 16) | (1 << 20) | (pad_iv << 22) | (pad_amt << 25); // 2B elems, pad on
  g1[1] = (dim0 & 0xFFFF) << 16;                                    // tensor_dim0[15:0]
  g1[2] = (int)(((unsigned)dim0 >> 16) | ((unsigned)(rows & 0xFFFF) << 16));
  g1[3] = (int)((((unsigned)rows >> 16) & 0xFFFFu) | ((unsigned)tile_d0 << 16));
  g1[4] = tile_rows & 0xFFFF;                                       // tile_dim1, tile_dim2=0
  g1[5] = stride0;                                                  // dim0_stride[31:0]
  g1[6] = 0;
  g1[7] = 0;
  tdm4i z4; z4[0] = 0; z4[1] = 0; z4[2] = 0; z4[3] = 0;
  tdm8i z8; for (int j = 0; j < 8; ++j) z8[j] = 0;
  // clang-23 / therock-10.0 form: 6 args (extra int32x8 group before cpol)
  __builtin_amdgcn_tensor_load_to_lds(g0, g1, z4, z4, z8, 0);
}
#endif

// ---------------------------------------------------------------------------
// Implicit-GEMM conv / transpose-conv. A = f16 weights [Cout][kgemm] row-major
// (deconv weights pre-flipped on host side into the same layout).
// Block: 128 threads / 4 waves; tile (16*mblocks) x 64 pixels.
// ---------------------------------------------------------------------------
__global__ __launch_bounds__(128) void conv_wmma(
    const _Float16* __restrict__ in, const _Float16* __restrict__ wA,
    const float* __restrict__ bias, const _Float16* __restrict__ residual,
    _Float16* __restrict__ out16, float* __restrict__ out32,
    int Cin, int Hin, int Win, int Cout, int Hout, int Wout,
    int K, int stride, int pad, int kgemm, int flags, int mblocks)
{
  __shared__ _Float16 Al[64][LPAD];
  __shared__ _Float16 Bl[64][LPAD];
  __shared__ int PBase[64], POy[64], POx[64], PNimg[64], PRem[64];
  __shared__ int Kci[32], Kky[32], Kkx[32];

  const int tid  = threadIdx.x;
  const int lane = tid & 31;
  const int wave = tid >> 5;
  const int HW   = Hout * Wout;
  const int HinWin = Hin * Win;
  const int n0   = blockIdx.x * 64;
  const int MT   = mblocks * 16;
  const int cout0 = blockIdx.y * MT;
  const int KK   = K * K;
  const int m_idx = wave % mblocks;
  const int ngrp  = wave / mblocks;
  const bool dec = (flags & FLAG_DECONV) != 0;
  const bool rin = (flags & FLAG_RELU_IN) != 0;

  if (tid < 64) {              // per-pixel tables: all divisions happen once
    int p = n0 + tid;
    int nimg = p / HW; int rem = p - nimg * HW;
    int oy = rem / Wout, ox = rem - oy * Wout;
    PBase[tid] = nimg * Cin * HinWin;
    POy[tid] = oy; POx[tid] = ox; PNimg[tid] = nimg; PRem[tid] = rem;
  }

  v8f acc[4];
  for (int s = 0; s < 4; ++s)
    for (int j = 0; j < 8; ++j) acc[s][j] = 0.0f;

  const int ksteps = (kgemm + 31) >> 5;
  for (int ks = 0; ks < ksteps; ++ks) {
    const int kbase = ks << 5;
    __syncthreads();                       // tiles/LUTs free for refill
#if USE_TDM
    if (wave == 0) {                       // DMA weight tile; overlaps B-fill
      tdm_load_2d_f16(wA + (size_t)cout0 * kgemm + kbase,
                      (unsigned)(uintptr_t)&Al[0][0],
                      kgemm - kbase, Cout - cout0, 32, MT, kgemm,
                      /*16 dwords*/3, /*pad 4 dwords*/3);
    }
#else
    for (int i = tid; i < MT * 32; i += 128) {
      int m = i >> 5, k = i & 31;
      int kg = kbase + k, mg = cout0 + m;
      Al[m][k] = (kg < kgemm && mg < Cout) ? wA[(size_t)mg * kgemm + kg]
                                           : (_Float16)0.0f;
    }
#endif
    if (tid < 32) {                        // per-k-step LUT: 32 divs total
      int kg = kbase + tid;
      int ci = -1, ky = 0, kx = 0;
      if (kg < kgemm) {
        ci = kg / KK; int r = kg - ci * KK; ky = r / K; kx = r - ky * K;
      }
      Kci[tid] = ci; Kky[tid] = ky; Kkx[tid] = kx;
    }
    __syncthreads();                       // LUTs visible (TDM still in flight)
    // ---- im2col B fill: adds/compares only -----------------------------
    for (int i = tid; i < 64 * 32; i += 128) {
      int n = i >> 5, k = i & 31;
      int ci = Kci[k];
      float v = 0.0f;
      if (ci >= 0) {
        int oy = POy[n], ox = POx[n];
        int ky = Kky[k], kx = Kkx[k];
        int iy, ix; bool ok;
        if (dec) {                         // k=4, s=2, p=1 transpose-conv
          int ty = oy + 1 - ky, tx = ox + 1 - kx;
          ok = ((ty | tx) >= 0) && ((ty & 1) == 0) && ((tx & 1) == 0);
          iy = ty >> 1; ix = tx >> 1;
          ok = ok && (iy < Hin) && (ix < Win);
        } else {
          iy = oy * stride - pad + ky; ix = ox * stride - pad + kx;
          ok = (iy >= 0) && (iy < Hin) && (ix >= 0) && (ix < Win);
        }
        if (ok) {
          v = (float)in[PBase[n] + ci * HinWin + iy * Win + ix];
          if (rin) v = fmaxf(v, 0.0f);
        }
      }
      Bl[n][k] = (_Float16)v;
    }
#if USE_TDM
    if (wave == 0) __builtin_amdgcn_s_wait_tensorcnt(0);
#endif
    __syncthreads();                       // Al (DMA) + Bl ready
    // ---- fragments + WMMA ---------------------------------------------
    const int mrow = m_idx * 16 + (lane & 15);
    const int kb   = (lane & 16) ? 8 : 0;
    v8h alo = *(const v8h*)&Al[mrow][kb];
    v8h ahi = *(const v8h*)&Al[mrow][16 + kb];
    v16h afrag = cat8(alo, ahi);
    const int kofB = (lane & 16) ? 16 : 0;
    for (int s = 0; s < mblocks; ++s) {
      int nrow = (ngrp * mblocks + s) * 16 + (lane & 15);
      v8h blo = *(const v8h*)&Bl[nrow][kb];
      v8h bhi = *(const v8h*)&Bl[nrow][16 + kb];
      acc[s] = __builtin_amdgcn_wmma_f32_16x16x32_f16(
          false, afrag, false, cat8(blo, bhi), (short)0, acc[s], false, false);
    }
  }

  // ---- epilogue: bias (+residual) (+relu/tanh), NCHW stores ------------
  for (int s = 0; s < mblocks; ++s) {
    int nn = (ngrp * mblocks + s) * 16 + (lane & 15);
    int nimg = PNimg[nn], rem = PRem[nn];
    int mbase = cout0 + m_idx * 16 + ((lane & 16) ? 8 : 0);
    for (int v = 0; v < 8; ++v) {
      int co = mbase + v;
      if (co < Cout) {
        float r = acc[s][v] + bias[co];
        int oidx = (nimg * Cout + co) * HW + rem;
        if (residual) r += (float)residual[oidx];
        if (flags & FLAG_RELU_OUT) r = fmaxf(r, 0.0f);
        if (flags & FLAG_TANH_OUT) r = tanhf(r);
        if (out16) out16[oidx] = (_Float16)r;
        if (out32) out32[oidx] = r;
      }
    }
  }
}

// ---------------------------------------------------------------------------
// Prep kernels: f32->f16 convert; deconv weight flip/transpose into
// [Cout][Cin*16] row-major f16 (so every layer's A-tile is TDM-friendly).
// ---------------------------------------------------------------------------
__global__ __launch_bounds__(256) void cvt_f16(const float* __restrict__ a,
                                               _Float16* __restrict__ o, int n) {
  int i = blockIdx.x * 256 + threadIdx.x;
  if (i < n) o[i] = (_Float16)a[i];
}

__global__ __launch_bounds__(256) void prep_deconv_w(const float* __restrict__ w,
                                                     _Float16* __restrict__ o,
                                                     int Cin, int Cout) {
  int i = blockIdx.x * 256 + threadIdx.x;      // over Cout*Cin*16
  int total = Cin * Cout * 16;
  if (i < total) {
    int co = i / (Cin * 16); int r = i - co * Cin * 16;
    int ci = r >> 4; int kk = r & 15;
    int ky = kk >> 2, kx = kk & 3;
    o[i] = (_Float16)w[((ci * Cout + co) * 4 + (3 - ky)) * 4 + (3 - kx)];
  }
}

__global__ __launch_bounds__(256) void cb_norm(const float* __restrict__ cb,
                                               float* __restrict__ cbn) {
  int e = blockIdx.x * 256 + threadIdx.x;
  if (e < 512) {
    float s = 0.0f;
    for (int d = 0; d < 64; ++d) { float v = cb[e * 64 + d]; s += v * v; }
    cbn[e] = s;
  }
}

// ---------------------------------------------------------------------------
// VQ argmin via WMMA: score[n][e] = ||E_e||^2 - 2 z_n.E_e ; codebook chunks
// DMA'd to LDS by TDM.  z is f16 NCHW (32,64,64,64).
// ---------------------------------------------------------------------------
__global__ __launch_bounds__(128) void vq_argmin(
    const _Float16* __restrict__ zh, const _Float16* __restrict__ cbh,
    const float* __restrict__ cbnorm, int* __restrict__ idx_out)
{
  __shared__ _Float16 Cb[16][CPAD];
  __shared__ _Float16 Zt[64][CPAD];

  const int tid = threadIdx.x, lane = tid & 31, wave = tid >> 5;
  const int n0 = blockIdx.x * 64;

  for (int i = tid; i < 64 * 64; i += 128) {
    int n = i & 63, d = i >> 6;
    int p = n0 + n;
    int nimg = p >> 12, yx = p & 4095;
    Zt[n][d] = zh[((size_t)nimg * 64 + d) * 4096 + yx];
  }

  float bestv = 3.4e38f; int besti = 0;
  const int nrow = wave * 16 + (lane & 15);
  const int kofB = (lane & 16) ? 16 : 0;
  const int kb   = (lane & 16) ? 8 : 0;
  const int mrow = lane & 15;

  for (int chunk = 0; chunk < 32; ++chunk) {
    __syncthreads();
#if USE_TDM
    if (wave == 0) {
      tdm_load_2d_f16(cbh + chunk * 16 * 64, (unsigned)(uintptr_t)&Cb[0][0],
                      64, 16, 64, 16, 64, /*32 dwords*/4, /*pad 4 dwords*/3);
      __builtin_amdgcn_s_wait_tensorcnt(0);
    }
#else
    for (int i = tid; i < 16 * 64; i += 128) {
      int e = i >> 6, d = i & 63;
      Cb[e][d] = cbh[(chunk * 16 + e) * 64 + d];
    }
#endif
    __syncthreads();
    v8f c; for (int j = 0; j < 8; ++j) c[j] = 0.0f;
    for (int ksv = 0; ksv < 2; ++ksv) {
      v8h alo = *(const v8h*)&Cb[mrow][ksv * 32 + kb];
      v8h ahi = *(const v8h*)&Cb[mrow][ksv * 32 + 16 + kb];
      v8h blo = *(const v8h*)&Zt[nrow][ksv * 32 + kb];
      v8h bhi = *(const v8h*)&Zt[nrow][ksv * 32 + 16 + kb];
      c = __builtin_amdgcn_wmma_f32_16x16x32_f16(
          false, cat8(alo, ahi), false, cat8(blo, bhi), (short)0, c, false, false);
    }
    int ebase = chunk * 16 + ((lane & 16) ? 8 : 0);
    for (int v = 0; v < 8; ++v) {            // ascending e, strict < = first min
      float score = cbnorm[ebase + v] - 2.0f * c[v];
      if (score < bestv) { bestv = score; besti = ebase + v; }
    }
  }
  float ov = __shfl_xor(bestv, 16, 32);
  int   oi = __shfl_xor(besti, 16, 32);
  if (lane < 16) {
    if (ov < bestv || (ov == bestv && oi < besti)) { bestv = ov; besti = oi; }
    idx_out[n0 + wave * 16 + lane] = besti;
  }
}

// ---------------------------------------------------------------------------
// Gather quantized = codebook[idx] (f16) + deterministic loss partials.
// Forward straight-through value is q; loss = 1.25 * mean((q-z)^2).
// ---------------------------------------------------------------------------
__global__ __launch_bounds__(256) void vq_gather_loss(
    const _Float16* __restrict__ zh, const float* __restrict__ cb,
    const int* __restrict__ idx, _Float16* __restrict__ q,
    float* __restrict__ part)
{
  float accum = 0.0f;
  for (int it = 0; it < 8; ++it) {
    int g = blockIdx.x * 2048 + it * 256 + threadIdx.x;  // < 8388608
    int d = (g >> 12) & 63;
    int nimg = g >> 18;
    int yx = g & 4095;
    int n = (nimg << 12) | yx;
    float qv = cb[idx[n] * 64 + d];
    float zv = (float)zh[g];
    q[g] = (_Float16)qv;
    float df = qv - zv;
    accum += df * df;
  }
  __shared__ float red[256];
  red[threadIdx.x] = accum;
  __syncthreads();
  for (int s = 128; s > 0; s >>= 1) {
    if (threadIdx.x < s) red[threadIdx.x] += red[threadIdx.x + s];
    __syncthreads();
  }
  if (threadIdx.x == 0) part[blockIdx.x] = red[0];
}

__global__ __launch_bounds__(256) void loss_final(const float* __restrict__ part,
                                                  float* __restrict__ out_loss) {
  __shared__ float red[256];
  float a = 0.0f;
  for (int i = threadIdx.x; i < 4096; i += 256) a += part[i];
  red[threadIdx.x] = a;
  __syncthreads();
  for (int s = 128; s > 0; s >>= 1) {
    if (threadIdx.x < s) red[threadIdx.x] += red[threadIdx.x + s];
    __syncthreads();
  }
  if (threadIdx.x == 0) out_loss[0] = 1.25f * red[0] / 8388608.0f;
}

// ---------------------------------------------------------------------------
// Host launch sequence
// ---------------------------------------------------------------------------
extern "C" void kernel_launch(void* const* d_in, const int* in_sizes, int n_in,
                              void* d_out, int out_size, void* d_ws, size_t ws_size,
                              hipStream_t stream) {
  (void)in_sizes; (void)n_in; (void)out_size; (void)ws_size;
  const float* x       = (const float*)d_in[0];
  const float* enc_w1  = (const float*)d_in[1];
  const float* enc_b1  = (const float*)d_in[2];
  const float* enc_w2  = (const float*)d_in[3];
  const float* enc_b2  = (const float*)d_in[4];
  const float* enc_w3  = (const float*)d_in[5];
  const float* enc_b3  = (const float*)d_in[6];
  const float* enc_rw1 = (const float*)d_in[7];
  const float* enc_rb1 = (const float*)d_in[8];
  const float* enc_rw2 = (const float*)d_in[9];
  const float* enc_rb2 = (const float*)d_in[10];
  const float* pvq_w   = (const float*)d_in[11];
  const float* pvq_b   = (const float*)d_in[12];
  const float* cbk     = (const float*)d_in[13];
  const float* dec_w1  = (const float*)d_in[14];
  const float* dec_b1  = (const float*)d_in[15];
  const float* dec_rw1 = (const float*)d_in[16];
  const float* dec_rb1 = (const float*)d_in[17];
  const float* dec_rw2 = (const float*)d_in[18];
  const float* dec_rb2 = (const float*)d_in[19];
  const float* dt1_w   = (const float*)d_in[20];
  const float* dt1_b   = (const float*)d_in[21];
  const float* dt2_w   = (const float*)d_in[22];
  const float* dt2_b   = (const float*)d_in[23];
  float* out = (float*)d_out;

  // ---- workspace carve-up (256B aligned) ----
  char* wsb = (char*)d_ws;
  auto a16 = [&](size_t n) { _Float16* p = (_Float16*)wsb; wsb += ((n * 2 + 255) & ~(size_t)255); return p; };
  auto a32 = [&](size_t n) { float* p = (float*)wsb; wsb += ((n * 4 + 255) & ~(size_t)255); return p; };
  _Float16* Xh   = a16((size_t)32 * 3 * 256 * 256);
  _Float16* A1h  = a16((size_t)32 * 64 * 128 * 128);
  _Float16* A2h  = a16((size_t)32 * 128 * 64 * 64);
  _Float16* A3h  = a16((size_t)32 * 128 * 64 * 64);
  _Float16* H32h = a16((size_t)32 * 32 * 64 * 64);
  _Float16* Zh   = a16((size_t)32 * 64 * 64 * 64);
  _Float16* Qh   = a16((size_t)32 * 64 * 64 * 64);
  _Float16* W1h  = a16(64 * 48);
  _Float16* W2h  = a16(128 * 1024);
  _Float16* W3h  = a16(128 * 1152);
  _Float16* ERW1h = a16(2 * 32 * 1152);
  _Float16* ERW2h = a16(2 * 128 * 32);
  _Float16* PVQh  = a16(64 * 128);
  _Float16* DW1h  = a16(128 * 576);
  _Float16* DRW1h = a16(2 * 32 * 1152);
  _Float16* DRW2h = a16(2 * 128 * 32);
  _Float16* DT1h  = a16(64 * 2048);
  _Float16* DT2h  = a16(3 * 1024);
  _Float16* CBh   = a16(512 * 64);
  float* PART = a32(4096);
  float* CBN  = a32(512);
  int*   IDX  = (int*)a32(131072);

  auto cvt = [&](const float* s, _Float16* d, int n) {
    cvt_f16<<<dim3((n + 255) / 256), dim3(256), 0, stream>>>(s, d, n);
  };
  // ---- prep: f16 conversions + deconv weight re-layout ----
  cvt(x, Xh, 32 * 3 * 256 * 256);
  cvt(enc_w1, W1h, 64 * 48);
  cvt(enc_w2, W2h, 128 * 1024);
  cvt(enc_w3, W3h, 128 * 1152);
  cvt(enc_rw1, ERW1h, 2 * 32 * 1152);
  cvt(enc_rw2, ERW2h, 2 * 128 * 32);
  cvt(pvq_w, PVQh, 64 * 128);
  cvt(dec_w1, DW1h, 128 * 576);
  cvt(dec_rw1, DRW1h, 2 * 32 * 1152);
  cvt(dec_rw2, DRW2h, 2 * 128 * 32);
  cvt(cbk, CBh, 512 * 64);
  prep_deconv_w<<<dim3((128 * 64 * 16 + 255) / 256), dim3(256), 0, stream>>>(dt1_w, DT1h, 128, 64);
  prep_deconv_w<<<dim3((64 * 3 * 16 + 255) / 256), dim3(256), 0, stream>>>(dt2_w, DT2h, 64, 3);

  dim3 blk(128);
  auto conv = [&](const _Float16* in, const _Float16* w, const float* b,
                  const _Float16* res, _Float16* o16, float* o32,
                  int Cin, int Hin, int Win, int Cout, int Hout, int Wout,
                  int K, int st, int pd, int flags, int mblocks) {
    int kg = Cin * K * K;
    int npx = 32 * Hout * Wout;
    dim3 grid(npx / 64, (Cout + 16 * mblocks - 1) / (16 * mblocks));
    conv_wmma<<<grid, blk, 0, stream>>>(in, w, b, res, o16, o32,
        Cin, Hin, Win, Cout, Hout, Wout, K, st, pd, kg, flags, mblocks);
  };

  // ---- encoder ----
  conv(Xh,  W1h, enc_b1, nullptr, A1h, nullptr, 3,  256, 256, 64,  128, 128, 4, 2, 1, FLAG_RELU_OUT, 4);
  conv(A1h, W2h, enc_b2, nullptr, A2h, nullptr, 64, 128, 128, 128, 64,  64,  4, 2, 1, FLAG_RELU_OUT, 4);
  conv(A2h, W3h, enc_b3, nullptr, A3h, nullptr, 128, 64, 64,  128, 64,  64,  3, 1, 1, 0, 4);
  conv(A3h,  ERW1h,        enc_rb1,       nullptr, H32h, nullptr, 128, 64, 64, 32,  64, 64, 3, 1, 1, FLAG_RELU_IN, 2);
  conv(H32h, ERW2h,        enc_rb2,       A3h,     A2h,  nullptr, 32,  64, 64, 128, 64, 64, 1, 1, 0, FLAG_RELU_IN, 4);
  conv(A2h,  ERW1h + 36864, enc_rb1 + 32, nullptr, H32h, nullptr, 128, 64, 64, 32,  64, 64, 3, 1, 1, FLAG_RELU_IN, 2);
  conv(H32h, ERW2h + 4096,  enc_rb2 + 128, A2h,    A3h,  nullptr, 32,  64, 64, 128, 64, 64, 1, 1, 0, FLAG_RELU_IN, 4);
  conv(A3h, PVQh, pvq_b, nullptr, Zh, nullptr, 128, 64, 64, 64, 64, 64, 1, 1, 0, FLAG_RELU_IN, 4);

  // ---- vector quantization ----
  cb_norm<<<dim3(2), dim3(256), 0, stream>>>(cbk, CBN);
  vq_argmin<<<dim3(131072 / 64), dim3(128), 0, stream>>>(Zh, CBh, CBN, IDX);
  vq_gather_loss<<<dim3(4096), dim3(256), 0, stream>>>(Zh, cbk, IDX, Qh, PART);
  loss_final<<<dim3(1), dim3(256), 0, stream>>>(PART, out + (size_t)32 * 3 * 256 * 256);

  // ---- decoder ----
  conv(Qh,   DW1h, dec_b1, nullptr, A2h, nullptr, 64, 64, 64, 128, 64, 64, 3, 1, 1, 0, 4);
  conv(A2h,  DRW1h,        dec_rb1,       nullptr, H32h, nullptr, 128, 64, 64, 32,  64, 64, 3, 1, 1, FLAG_RELU_IN, 2);
  conv(H32h, DRW2h,        dec_rb2,       A2h,     A3h,  nullptr, 32,  64, 64, 128, 64, 64, 1, 1, 0, FLAG_RELU_IN, 4);
  conv(A3h,  DRW1h + 36864, dec_rb1 + 32, nullptr, H32h, nullptr, 128, 64, 64, 32,  64, 64, 3, 1, 1, FLAG_RELU_IN, 2);
  conv(H32h, DRW2h + 4096,  dec_rb2 + 128, A3h,    A2h,  nullptr, 32,  64, 64, 128, 64, 64, 1, 1, 0, FLAG_RELU_IN, 4);
  conv(A2h, DT1h, dt1_b, nullptr, A1h, nullptr, 128, 64, 64, 64, 128, 128, 4, 2, 1,
       FLAG_RELU_IN | FLAG_RELU_OUT | FLAG_DECONV, 4);
  conv(A1h, DT2h, dt2_b, nullptr, nullptr, out, 64, 128, 128, 3, 256, 256, 4, 2, 1,
       FLAG_TANH_OUT | FLAG_DECONV, 1);
}
